// Seq2seq_model_53154515256309
// MI455X (gfx1250) — matrix operand order repaired and checked by reading, not compiled, not run on hardware
//
#include <hip/hip_runtime.h>
#include <hip/hip_bf16.h>

// ---------------- problem constants (match reference) ----------------
#define B_  2048
#define S_  512
#define T_  96
#define H_  16

typedef _Float16 v16h __attribute__((ext_vector_type(16)));
typedef float    v8f  __attribute__((ext_vector_type(8)));

union BU { v16h v; unsigned u[8]; };

// pack two f32 -> two f16 in one dword
static __device__ __forceinline__ unsigned pkh2(float a, float b) {
    union { _Float16 h[2]; unsigned u; } t;
    t.h[0] = (_Float16)a; t.h[1] = (_Float16)b;
    return t.u;
}

// CDNA5 transcendental tanh (gfx1250 VOP1 trans op)
static __device__ __forceinline__ float tanh_fast(float x) {
    float r;
    asm("v_tanh_f32 %0, %1" : "=v"(r) : "v"(x));
    return r;
}
static __device__ __forceinline__ float sigm(float x) {
    return 0.5f * tanh_fast(0.5f * x) + 0.5f;   // sigmoid via one tanh
}

static __device__ __forceinline__ v8f wmma16(v16h A, v16h B, v8f C) {
    // (neg_a, A, neg_b, B, c_mod, C, reuse_a, reuse_b)
    return __builtin_amdgcn_wmma_f32_16x16x32_f16(false, A, false, B, (short)0, C,
                                                  false, false);
}

static __device__ __forceinline__ uint4 swap16(uint4 v) {
    uint4 r;
    r.x = __shfl_xor(v.x, 16, 32);
    r.y = __shfl_xor(v.y, 16, 32);
    r.z = __shfl_xor(v.z, 16, 32);
    r.w = __shfl_xor(v.w, 16, 32);
    return r;
}

// One LSTM layer's parameters, pre-arranged in WMMA operand layouts.
// A[q]  : gate chunk q (16 gates x 32 K) in the f16 A-matrix layout
// Cb[q] : bias(bi+bh) broadcast in the f32 C/D layout (used as WMMA C input)
struct Layer {
    v16h A[4];
    v8f  Cb[4];
};

// Hidden state kept as packed f16 (pk) PLUS its lane-half-swapped copy (spk),
// so consumers never need to re-shuffle on the serial critical path.
//  pk : lane<16 -> units 0..7 of batch b0+lane ; lane>=16 -> units 8..15
//  spk: partner lane's pk (shfl_xor 16), refreshed once per update
struct HState {
    uint4 pk;
    uint4 spk;
};

// Load W_cat = [Wi | Wh] (64 x 32, K 0..15 = input, K 16..31 = hidden) into
// the 16x32 f16 A layout:
//   lanes 0-15  row M=lane   : halfs 0..7 = K 0..7,  halfs 8..15 = K 16..23
//   lanes 16-31 row M=lane-16: halfs 0..7 = K 8..15, halfs 8..15 = K 24..31
static __device__ __forceinline__ void load_layer(Layer& L,
        const float* __restrict__ Wi, const float* __restrict__ Wh,
        const float* __restrict__ bi, const float* __restrict__ bh,
        int in_dim, int lane)
{
    const int  m   = lane & 15;
    const bool lo  = lane < 16;
    const int  kin = lo ? 0 : 8;   // input-part K base for halfs 0..7
    const int  kh  = lo ? 0 : 8;   // hidden-part col base for halfs 8..15
#pragma unroll
    for (int q = 0; q < 4; ++q) {
        const int g = q * 16 + m;
        v16h a;
#pragma unroll
        for (int e = 0; e < 8; ++e) {
            const int k = kin + e;
            const float w = (k < in_dim) ? Wi[g * in_dim + k] : 0.0f;
            a[e] = (_Float16)w;
        }
#pragma unroll
        for (int e = 0; e < 8; ++e)
            a[8 + e] = (_Float16)Wh[g * H_ + kh + e];
        L.A[q] = a;

        // C/D layout: lanes 0-15 rows 0..7, lanes 16-31 rows 8..15
        v8f cb;
#pragma unroll
        for (int e = 0; e < 8; ++e) {
            const int gg = q * 16 + (lo ? e : 8 + e);
            cb[e] = bi[gg] + bh[gg];
        }
        L.Cb[q] = cb;
    }
}

// One LSTM cell step for 16 batch columns on one wave.
//  din_pk / din_spk : layer input (packed f16) and its swapped copy
//  h : hidden state (pk + swapped copy), updated in place (4 bpermutes)
//  c : cell state in the f32 D layout, updated in place
//  hv: new hidden (f32, D layout) out
static __device__ __forceinline__ void lstm_cell(const Layer& L,
        uint4 din_pk, uint4 din_spk, HState& h, float c[8], float hv[8], bool lo)
{
    // Build the 32x16 f16 B operand:
    //  lanes 0-15  hold K 0..7 (own din) and K 16..23 (own h)
    //  lanes 16-31 hold K 8..15 (own din) and K 24..31 (own h)
    BU Bop;
    Bop.u[0] = din_pk.x;
    Bop.u[1] = din_pk.y;
    Bop.u[2] = din_pk.z;
    Bop.u[3] = din_pk.w;
    Bop.u[4] = h.pk.x;
    Bop.u[5] = h.pk.y;
    Bop.u[6] = h.pk.z;
    Bop.u[7] = h.pk.w;

    // z = W_cat * [x;h] + b, gate chunks in torch order i,f,g,o
    const v8f zi = wmma16(L.A[0], Bop.v, L.Cb[0]);
    const v8f zf = wmma16(L.A[1], Bop.v, L.Cb[1]);
    const v8f zg = wmma16(L.A[2], Bop.v, L.Cb[2]);
    const v8f zo = wmma16(L.A[3], Bop.v, L.Cb[3]);

#pragma unroll
    for (int e = 0; e < 8; ++e) {
        const float cn = sigm(zf[e]) * c[e] + sigm(zi[e]) * tanh_fast(zg[e]);
        c[e]  = cn;
        hv[e] = sigm(zo[e]) * tanh_fast(cn);
    }
    h.pk.x = pkh2(hv[0], hv[1]);
    h.pk.y = pkh2(hv[2], hv[3]);
    h.pk.z = pkh2(hv[4], hv[5]);
    h.pk.w = pkh2(hv[6], hv[7]);
    h.spk  = swap16(h.pk);             // the only cross-lane traffic per cell
}

__global__ __launch_bounds__(128, 1)
void seq2seq_lstm_kernel(
    const float* __restrict__ x,    const float* __restrict__ h0,
    const float* __restrict__ c0,
    const float* __restrict__ embW, const float* __restrict__ embB,
    const float* __restrict__ outW, const float* __restrict__ outB,
    const float* __restrict__ eW0i, const float* __restrict__ eW0h,
    const float* __restrict__ eb0i, const float* __restrict__ eb0h,
    const float* __restrict__ eW1i, const float* __restrict__ eW1h,
    const float* __restrict__ eb1i, const float* __restrict__ eb1h,
    const float* __restrict__ dW0i, const float* __restrict__ dW0h,
    const float* __restrict__ db0i, const float* __restrict__ db0h,
    const float* __restrict__ dW1i, const float* __restrict__ dW1h,
    const float* __restrict__ db1i, const float* __restrict__ db1h,
    float* __restrict__ out)
{
    const int  lane = threadIdx.x & 31;
    const int  wave = threadIdx.x >> 5;
    const int  tile = blockIdx.x * 4 + wave;       // 0..127
    const int  b0   = tile * 16;
    const bool lo   = lane < 16;
    const int  b    = b0 + (lane & 15);            // this lane's batch column
    const int  ub   = lo ? 0 : 8;                  // unit base for this half

    // ---- state init from h0/c0 ----
    float c0s[8], c1s[8], hv0[8], hv1[8];
    HState h0s, h1s;
    {
        const float* h0p = h0 + b * H_ + ub;
        const float* c0p = c0 + b * H_ + ub;
        const float* h1p = h0 + B_ * H_ + b * H_ + ub;
        const float* c1p = c0 + B_ * H_ + b * H_ + ub;
        float t0[8], t1[8];
#pragma unroll
        for (int e = 0; e < 8; ++e) {
            c0s[e] = c0p[e]; c1s[e] = c1p[e];
            t0[e]  = h0p[e]; t1[e]  = h1p[e];
        }
        h0s.pk.x = pkh2(t0[0], t0[1]); h0s.pk.y = pkh2(t0[2], t0[3]);
        h0s.pk.z = pkh2(t0[4], t0[5]); h0s.pk.w = pkh2(t0[6], t0[7]);
        h1s.pk.x = pkh2(t1[0], t1[1]); h1s.pk.y = pkh2(t1[2], t1[3]);
        h1s.pk.z = pkh2(t1[4], t1[5]); h1s.pk.w = pkh2(t1[6], t1[7]);
        h0s.spk = swap16(h0s.pk);
        h1s.spk = swap16(h1s.pk);
    }

    float last_x = 0.f, last_d1 = 0.f, last_d2 = 0.f;
    const uint4 zero4 = make_uint4(0u, 0u, 0u, 0u);

    // ================= encoder: 512 steps of 2-layer LSTM =================
    {
        Layer L0, L1;
        load_layer(L0, eW0i, eW0h, eb0i, eb0h, 3,  lane);
        load_layer(L1, eW1i, eW1h, eb1i, eb1h, 16, lane);

        const float* px = x + b * S_;              // this lane's time series
        float xprev  = px[0];                      // makes d1==0 at t==0
        float d1prev = 0.f;

        for (int t4 = 0; t4 < S_; t4 += 4) {
            const float4 xq = *(const float4*)(px + t4);
            const float xs[4] = { xq.x, xq.y, xq.z, xq.w };
#pragma unroll
            for (int j = 0; j < 4; ++j) {
                const float xv = xs[j];
                const float d1 = xv - xprev;
                const float d2 = d1 - d1prev;
                xprev = xv; d1prev = d1;
                last_x = xv; last_d1 = d1; last_d2 = d2;

                // layer-0 input: [x, d1, d2, 0...0] padded to K=16.
                // Lanes >=16 carry the zero padding, so the swapped copy of
                // this input is identically zero -> no shuffle needed.
                uint4 dpk;
                dpk.x = lo ? pkh2(xv, d1)  : 0u;
                dpk.y = lo ? pkh2(d2, 0.f) : 0u;
                dpk.z = 0u; dpk.w = 0u;

                lstm_cell(L0, dpk,     zero4,    h0s, c0s, hv0, lo);
                lstm_cell(L1, h0s.pk,  h0s.spk,  h1s, c1s, hv1, lo);
            }
        }
    }

    // ================= decoder: 96 autoregressive steps =================
    {
        Layer L0, L1;
        load_layer(L0, dW0i, dW0h, db0i, db0h, 16, lane);
        load_layer(L1, dW1i, dW1h, db1i, db1h, 16, lane);

        float ow[8];
#pragma unroll
        for (int e = 0; e < 8; ++e) ow[e] = outW[ub + e];
        const float ob = outB[0];

        // dec_in = emb(xin[:, -1, :]) : lane-local (every lane holds its
        // batch's last (x,d1,d2) since both wave halves streamed the row)
        uint4 dpk, dspk;
        {
            float di[8];
#pragma unroll
            for (int e = 0; e < 8; ++e) {
                const int u = ub + e;
                di[e] = embW[u * 3 + 0] * last_x
                      + embW[u * 3 + 1] * last_d1
                      + embW[u * 3 + 2] * last_d2
                      + embB[u];
            }
            dpk.x = pkh2(di[0], di[1]); dpk.y = pkh2(di[2], di[3]);
            dpk.z = pkh2(di[4], di[5]); dpk.w = pkh2(di[6], di[7]);
            dspk  = swap16(dpk);
        }

        float* po = out + b * T_;
        for (int t = 0; t < T_; ++t) {
            lstm_cell(L0, dpk,    dspk,     h0s, c0s, hv0, lo);
            lstm_cell(L1, h0s.pk, h0s.spk,  h1s, c1s, hv1, lo);

            // y = out_W . h1 + out_b  (reduce 16 units across the lane pair)
            float p = 0.f;
#pragma unroll
            for (int e = 0; e < 8; ++e) p += ow[e] * hv1[e];
            p += __shfl_xor(p, 16, 32);
            if (lo) po[t] = p + ob;

            // feedback: next input = h1 -- pure register copy, no shuffles
            dpk  = h1s.pk;
            dspk = h1s.spk;
        }
    }
}

extern "C" void kernel_launch(void* const* d_in, const int* in_sizes, int n_in,
                              void* d_out, int out_size, void* d_ws, size_t ws_size,
                              hipStream_t stream) {
    (void)in_sizes; (void)n_in; (void)out_size; (void)d_ws; (void)ws_size;
    const float* x    = (const float*)d_in[0];
    const float* h0   = (const float*)d_in[1];
    const float* c0   = (const float*)d_in[2];
    const float* embW = (const float*)d_in[3];
    const float* embB = (const float*)d_in[4];
    const float* outW = (const float*)d_in[5];
    const float* outB = (const float*)d_in[6];
    const float* eW0i = (const float*)d_in[7];
    const float* eW0h = (const float*)d_in[8];
    const float* eb0i = (const float*)d_in[9];
    const float* eb0h = (const float*)d_in[10];
    const float* eW1i = (const float*)d_in[11];
    const float* eW1h = (const float*)d_in[12];
    const float* eb1i = (const float*)d_in[13];
    const float* eb1h = (const float*)d_in[14];
    const float* dW0i = (const float*)d_in[15];
    const float* dW0h = (const float*)d_in[16];
    const float* db0i = (const float*)d_in[17];
    const float* db0h = (const float*)d_in[18];
    const float* dW1i = (const float*)d_in[19];
    const float* dW1h = (const float*)d_in[20];
    const float* db1i = (const float*)d_in[21];
    const float* db1h = (const float*)d_in[22];

    // 128 waves total: 16 batch columns per wave, 4 waves/block, 32 blocks
    seq2seq_lstm_kernel<<<dim3(B_ / 64), dim3(128), 0, stream>>>(
        x, h0, c0, embW, embB, outW, outB,
        eW0i, eW0h, eb0i, eb0h, eW1i, eW1h, eb1i, eb1h,
        dW0i, dW0h, db0i, db0h, dW1i, dW1h, db1i, db1h,
        (float*)d_out);
}
